// MPNNLayer_21088289423715
// MI455X (gfx1250) — matrix operand order repaired and checked by reading, not compiled, not run on hardware
//
#include <hip/hip_runtime.h>

// ---------------- CDNA5 WMMA types ----------------
typedef __attribute__((ext_vector_type(16))) __bf16 v16bf;
typedef __attribute__((ext_vector_type(8)))  float  v8f;

union Frag16 { v16bf v; uint4 q[2]; };

__device__ __forceinline__ v16bf load_frag_lds(const unsigned short* p0,
                                               const unsigned short* p1) {
  Frag16 f;
  f.q[0] = *(const uint4*)p0;   // ds_load_b128
  f.q[1] = *(const uint4*)p1;   // ds_load_b128
  return f.v;
}

__device__ __forceinline__ unsigned int f2bf(float x) {
  unsigned int u = __float_as_uint(x);
  unsigned int r = u + 0x7FFFu + ((u >> 16) & 1u);   // round-to-nearest-even
  return r >> 16;
}

__device__ __forceinline__ float gelu_tanh(float x) {
  float x3 = x * x * x;
  float t  = tanhf(0.7978845608028654f * (x + 0.044715f * x3));
  return 0.5f * x * (1.0f + t);
}

// ---------------- kernel 0: zero the aggregation buffer ----------------
__global__ void zero_kernel(float4* __restrict__ p, int n4) {
  int i = blockIdx.x * blockDim.x + threadIdx.x;
  if (i < n4) p[i] = make_float4(0.f, 0.f, 0.f, 0.f);
}

// ---------------- kernel 1: edge messages + scatter-add ----------------
// messages = gelu([h[src] | e] @ W_msg + b_msg); agg[tgt] += messages
#define D       64
#define KMSG    128
#define TILE_E  128     // edges per block (8 waves x 16 rows)

__global__ __launch_bounds__(256) void msg_kernel(
    const float* __restrict__ nodef, const float* __restrict__ edgef,
    const float* __restrict__ Wm,    const float* __restrict__ bm,
    const int*   __restrict__ src,   const int* __restrict__ tgt,
    float* __restrict__ agg, int E)
{
  __shared__ unsigned short Wb[D * KMSG];      // [n][k] bf16, 16 KB
  __shared__ unsigned short Ab[TILE_E * KMSG]; // [edge][k] bf16, 32 KB
  __shared__ int   tgts[TILE_E];
  __shared__ float bsh[D];

  const int tid  = threadIdx.x;
  const int base = blockIdx.x * TILE_E;

  // W_msg is [K=128][N=64] row-major in memory -> LDS as [n][k] bf16, packed pairs
  for (int j = tid; j < D * (KMSG / 2); j += 256) {
    int n = j >> 6, kp = j & 63, k = kp * 2;
    unsigned int val = f2bf(Wm[(size_t)k * D + n]) |
                       (f2bf(Wm[(size_t)(k + 1) * D + n]) << 16);
    ((unsigned int*)Wb)[(size_t)n * (KMSG / 2) + kp] = val;
  }
  if (tid < D) bsh[tid] = bm[tid];
  if (tid < TILE_E) {
    int e = base + tid;
    tgts[tid] = (e < E) ? tgt[e] : 0;
  }
  // A tile: row r = edge, k<64 -> node_features[src[e]], k>=64 -> edge_features[e]
  for (int idx = tid; idx < TILE_E * (KMSG / 2); idx += 256) {
    int r = idx >> 6, kp = idx & 63, k = kp * 2;
    int e = base + r;
    unsigned int val = 0;
    if (e < E) {
      const float2* p = (kp < 32)
          ? (const float2*)(nodef + (size_t)src[e] * D + k)
          : (const float2*)(edgef + (size_t)e * D + (k - D));
      float2 f = *p;
      val = f2bf(f.x) | (f2bf(f.y) << 16);
    }
    ((unsigned int*)Ab)[idx] = val;
  }
  __syncthreads();

  const int wave = tid >> 5;      // wave32
  const int lane = tid & 31;
  const int hh   = lane >> 4;     // lane half
  const int mrow = lane & 15;
  const int erow = wave * 16;     // this wave's 16 edges

  v8f acc[4];
#pragma unroll
  for (int t = 0; t < 4; ++t)
#pragma unroll
    for (int v = 0; v < 8; ++v) acc[t][v] = 0.0f;

#pragma unroll
  for (int s = 0; s < 4; ++s) {                       // K steps of 32
    const unsigned short* ar = &Ab[(size_t)(erow + mrow) * KMSG + s * 32 + hh * 8];
    v16bf a = load_frag_lds(ar, ar + 16);
#pragma unroll
    for (int t = 0; t < 4; ++t) {                     // N tiles of 16
      const unsigned short* br = &Wb[(size_t)(t * 16 + mrow) * KMSG + s * 32 + hh * 8];
      v16bf b = load_frag_lds(br, br + 16);
      acc[t] = __builtin_amdgcn_wmma_f32_16x16x32_bf16(
          false, a, false, b, (short)0, acc[t], false, false);
    }
  }

  // epilogue: bias + gelu + atomic scatter-add
#pragma unroll
  for (int t = 0; t < 4; ++t) {
    int n = t * 16 + mrow;
    float bias = bsh[n];
#pragma unroll
    for (int v = 0; v < 8; ++v) {
      int m = v + hh * 8;            // row within wave tile
      int e = base + erow + m;
      if (e < E) {
        float val = gelu_tanh(acc[t][v] + bias);
        atomicAdd(&agg[(size_t)tgts[erow + m] * D + n], val);
      }
    }
  }
}

// ---------------- kernel 2: node update ----------------
// out = gelu(((1+eps)*h + agg) @ W_upd + b_upd)
#define TILE_N 128

__global__ __launch_bounds__(256) void upd_kernel(
    const float* __restrict__ nodef, const float* __restrict__ agg,
    const float* __restrict__ Wu,    const float* __restrict__ bu,
    const float* __restrict__ epsp,  float* __restrict__ out, int Nn)
{
  __shared__ unsigned short Wb[D * D];       // [n][k] bf16, 8 KB
  __shared__ unsigned short Hb[TILE_N * D];  // [node][k] bf16, 16 KB
  __shared__ float bsh[D];

  const int tid  = threadIdx.x;
  const int base = blockIdx.x * TILE_N;
  const float e1 = 1.0f + epsp[0];

  for (int j = tid; j < D * (D / 2); j += 256) {
    int n = j >> 5, kp = j & 31, k = kp * 2;
    unsigned int val = f2bf(Wu[(size_t)k * D + n]) |
                       (f2bf(Wu[(size_t)(k + 1) * D + n]) << 16);
    ((unsigned int*)Wb)[(size_t)n * (D / 2) + kp] = val;
  }
  if (tid < D) bsh[tid] = bu[tid];

  for (int idx = tid; idx < TILE_N * (D / 2); idx += 256) {
    int r = idx >> 5, kp = idx & 31, k = kp * 2;
    int node = base + r;
    unsigned int val = 0;
    if (node < Nn) {
      float2 fn = *(const float2*)(nodef + (size_t)node * D + k);
      float2 fa = *(const float2*)(agg   + (size_t)node * D + k);
      float hx = e1 * fn.x + fa.x;
      float hy = e1 * fn.y + fa.y;
      val = f2bf(hx) | (f2bf(hy) << 16);
    }
    ((unsigned int*)Hb)[idx] = val;
  }
  __syncthreads();

  const int wave = tid >> 5;
  const int lane = tid & 31;
  const int hh   = lane >> 4;
  const int mrow = lane & 15;
  const int rrow = wave * 16;

  v8f acc[4];
#pragma unroll
  for (int t = 0; t < 4; ++t)
#pragma unroll
    for (int v = 0; v < 8; ++v) acc[t][v] = 0.0f;

#pragma unroll
  for (int s = 0; s < 2; ++s) {                       // K=64 -> 2 steps
    const unsigned short* ar = &Hb[(size_t)(rrow + mrow) * D + s * 32 + hh * 8];
    v16bf a = load_frag_lds(ar, ar + 16);
#pragma unroll
    for (int t = 0; t < 4; ++t) {
      const unsigned short* br = &Wb[(size_t)(t * 16 + mrow) * D + s * 32 + hh * 8];
      v16bf b = load_frag_lds(br, br + 16);
      acc[t] = __builtin_amdgcn_wmma_f32_16x16x32_bf16(
          false, a, false, b, (short)0, acc[t], false, false);
    }
  }

#pragma unroll
  for (int t = 0; t < 4; ++t) {
    int n = t * 16 + mrow;
    float bias = bsh[n];
#pragma unroll
    for (int v = 0; v < 8; ++v) {
      int m = v + hh * 8;
      int node = base + rrow + m;
      if (node < Nn)
        out[(size_t)node * D + n] = gelu_tanh(acc[t][v] + bias);
    }
  }
}

// ---------------- launcher ----------------
extern "C" void kernel_launch(void* const* d_in, const int* in_sizes, int n_in,
                              void* d_out, int out_size, void* d_ws, size_t ws_size,
                              hipStream_t stream) {
  const float* nodef = (const float*)d_in[0];
  const float* edgef = (const float*)d_in[1];
  const float* Wm    = (const float*)d_in[2];
  const float* bm    = (const float*)d_in[3];
  const float* Wu    = (const float*)d_in[4];
  const float* bu    = (const float*)d_in[5];
  const float* eps   = (const float*)d_in[6];
  const int*   src   = (const int*)d_in[7];
  const int*   tgt   = (const int*)d_in[8];

  const int Nn = in_sizes[0] / D;     // 100000
  const int E  = in_sizes[7];         // 1000000

  float* agg = (float*)d_ws;          // [Nn][64] fp32 scratch

  int n4 = (Nn * D) / 4;
  zero_kernel<<<(n4 + 255) / 256, 256, 0, stream>>>((float4*)agg, n4);

  int nblk = (E + TILE_E - 1) / TILE_E;
  msg_kernel<<<nblk, 256, 0, stream>>>(nodef, edgef, Wm, bm, src, tgt, agg, E);

  int ublk = (Nn + TILE_N - 1) / TILE_N;
  upd_kernel<<<ublk, 256, 0, stream>>>(nodef, agg, Wu, bu, eps, (float*)d_out, Nn);
}
